// Attention_61117384622766
// MI455X (gfx1250) — matrix operand order repaired and checked by reading, not compiled, not run on hardware
//
#include <hip/hip_runtime.h>
#include <hip/hip_bf16.h>

// ---------------- CDNA5 WMMA helpers (wave32, 16x16x32 f16 -> f32) ----------

typedef __attribute__((ext_vector_type(16))) _Float16 v16h;
typedef __attribute__((ext_vector_type(8)))  _Float16 v8h;
typedef __attribute__((ext_vector_type(8)))  float    v8f;
typedef int i32x4 __attribute__((vector_size(16)));   // matches async-LDS builtin param

// A-matrix 16x32 f16 fragment: lane l holds row M = l&15; kHalf = l>>4.
// element e -> K = kHalf*16 + e  => two adjacent 8-half chunks.
__device__ __forceinline__ v16h load_a_frag(const _Float16* row, int c0, int c1) {
  v8h lo = *(const v8h*)(row + c0);
  v8h hi = *(const v8h*)(row + c1);
  v16h r;
#pragma unroll
  for (int i = 0; i < 8; ++i) { r[i] = lo[i]; r[i + 8] = hi[i]; }
  return r;
}

// B-matrix 32x16 f16 fragment: lane l holds column N = l&15;
// element e -> K = (l>>4)*16 + e  => one contiguous 16-half (32B) chunk.
__device__ __forceinline__ v16h load_b_frag(const _Float16* p) {
  return *(const v16h*)p;
}

__device__ __forceinline__ v8f wmma16(v16h a, v16h b, v8f c) {
  return __builtin_amdgcn_wmma_f32_16x16x32_f16(
      /*neg_a=*/false, a, /*neg_b=*/false, b,
      /*c_mod=*/(short)0, c, /*reuse_a=*/false, /*reuse_b=*/false);
}

// Async global->LDS 16B copy (CDNA5 GLOBAL_LOAD_ASYNC_TO_LDS_B128, ASYNCcnt),
// with a synchronous fallback if this toolchain lacks the builtin.
#if defined(__has_builtin) && __has_builtin(__builtin_amdgcn_global_load_async_to_lds_b128)
#define HAVE_ASYNC_LDS 1
#else
#define HAVE_ASYNC_LDS 0
#endif

__device__ __forceinline__ void copy16_g2l(_Float16* ldst, const _Float16* gsrc) {
#if HAVE_ASYNC_LDS
  __builtin_amdgcn_global_load_async_to_lds_b128(
      (i32x4 __attribute__((address_space(1)))*)gsrc,
      (i32x4 __attribute__((address_space(3)))*)ldst,
      /*imm offset=*/0, /*cpol=*/0);
#else
  *(v8h*)ldst = *(const v8h*)gsrc;
#endif
}

__device__ __forceinline__ void wait_async_copies() {
#if HAVE_ASYNC_LDS
#if __has_builtin(__builtin_amdgcn_s_wait_asynccnt)
  __builtin_amdgcn_s_wait_asynccnt(0);
#else
  asm volatile("s_wait_asynccnt 0" ::: "memory");
#endif
#endif
}

// Problem constants
#define BATCH 4
#define SEQ   2048
#define DMODEL 512
#define HEADS 8
#define DHEAD 64
#define ROWS (BATCH * SEQ)        // 8192
#define NQKV (3 * DMODEL)         // 1536
#define KBLK 64                   // keys per attention tile

// ---------------- conversion / transpose kernels ----------------------------

__global__ void cvt_f32_to_f16(const float* __restrict__ in,
                               _Float16* __restrict__ out, int n) {
  int i = blockIdx.x * blockDim.x + threadIdx.x;
  if (i < n) out[i] = (_Float16)in[i];
}

// in[rows][cols] (row-major f32) -> out[cols][rows] (row-major f16)
__global__ void transpose_cvt(const float* __restrict__ in,
                              _Float16* __restrict__ out, int rows, int cols) {
  int i = blockIdx.x * blockDim.x + threadIdx.x;
  if (i < rows * cols) {
    int r = i / cols, c = i % cols;
    out[(size_t)c * rows + r] = (_Float16)in[i];
  }
}

// ---------------- QKV projection GEMM ---------------------------------------
// xh [8192,512] f16, wT [1536,512] f16 (column-major weights).
// Writes Q,K as [bh][n][64], V as [bh][64][n] (transposed for PV matmul).
__global__ void qkv_gemm(const _Float16* __restrict__ xh,
                         const _Float16* __restrict__ wT,
                         _Float16* __restrict__ qws,
                         _Float16* __restrict__ kws,
                         _Float16* __restrict__ vws) {
  const int lane = threadIdx.x & 31;
  const int wave = threadIdx.x >> 5;
  const int half = lane >> 4;
  const int lidx = lane & 15;
  const int mBase = blockIdx.x * 128 + wave * 16;  // token-row tile
  const int nBase = blockIdx.y * 64;               // output-col tile

  v8f acc[4];
#pragma unroll
  for (int t = 0; t < 4; ++t)
#pragma unroll
    for (int r = 0; r < 8; ++r) acc[t][r] = 0.0f;

  const _Float16* arow = xh + (size_t)(mBase + lidx) * DMODEL;
  for (int k0 = 0; k0 < DMODEL; k0 += 32) {
    v16h a = load_a_frag(arow, k0 + half * 16, k0 + half * 16 + 8);
#pragma unroll
    for (int t = 0; t < 4; ++t) {
      v16h bf = load_b_frag(wT + (size_t)(nBase + t * 16 + lidx) * DMODEL +
                            k0 + half * 16);
      acc[t] = wmma16(a, bf, acc[t]);
    }
  }

  // C layout: elem r -> row M = r + 8*(lane>>4), col N = lane&15
#pragma unroll
  for (int t = 0; t < 4; ++t) {
#pragma unroll
    for (int r = 0; r < 8; ++r) {
      int row = mBase + r + half * 8;      // 0..8191
      int col = nBase + t * 16 + lidx;     // 0..1535
      int bb = row >> 11;                  // batch
      int n  = row & 2047;
      int which = col >> 9;                // 0=Q 1=K 2=V (uniform per t)
      int w = col & 511;
      int h = w >> 6;
      int d = w & 63;
      _Float16 val = (_Float16)acc[t][r];
      size_t bh = (size_t)(bb * HEADS + h);
      if (which == 0)      qws[(bh * SEQ + n) * DHEAD + d] = val;
      else if (which == 1) kws[(bh * SEQ + n) * DHEAD + d] = val;
      else                 vws[(bh * DHEAD + d) * SEQ + n] = val;
    }
  }
}

// ---------------- flash attention core --------------------------------------
// One wave = 16 query rows; 8 waves/WG = 128 rows. Grid = 32 bh * 16 qblk.
// K/V tiles for the shared (b,h) stream are staged once per WG into LDS via
// async global->LDS copies; waves consume B-fragments from LDS (ds_load_b128).
__global__ void __launch_bounds__(256)
attn_kernel(const _Float16* __restrict__ qws,
            const _Float16* __restrict__ kws,
            const _Float16* __restrict__ vws,
            _Float16* __restrict__ oh) {
  __shared__ __attribute__((aligned(64))) _Float16 ldsK[KBLK * DHEAD];      // 8 KB
  __shared__ __attribute__((aligned(64))) _Float16 ldsV[DHEAD * KBLK];      // 8 KB
  __shared__ __attribute__((aligned(64))) _Float16 ldsP[8 * 16 * KBLK];     // 16 KB

  const int tid  = threadIdx.x;
  const int lane = tid & 31;
  const int wave = tid >> 5;
  const int half = lane >> 4;
  const int lidx = lane & 15;
  const int bh   = blockIdx.x >> 4;        // 0..31
  const int qblk = blockIdx.x & 15;
  const int qBase = qblk * 128 + wave * 16;
  const int bb = bh >> 3, hh = bh & 7;

  const _Float16* qb = qws + (size_t)bh * SEQ * DHEAD;
  const _Float16* kb = kws + (size_t)bh * SEQ * DHEAD;
  const _Float16* vb = vws + (size_t)bh * DHEAD * SEQ;

  // Q fragments (Dh=64 => two 16x32 A fragments), loaded once from global.
  const _Float16* qrow = qb + (size_t)(qBase + lidx) * DHEAD;
  v16h qf[2];
#pragma unroll
  for (int kt = 0; kt < 2; ++kt)
    qf[kt] = load_a_frag(qrow, kt * 32 + half * 16, kt * 32 + half * 16 + 8);

  v8f o[4];
  float m[8], l[8];
#pragma unroll
  for (int t = 0; t < 4; ++t)
#pragma unroll
    for (int r = 0; r < 8; ++r) o[t][r] = 0.0f;
#pragma unroll
  for (int r = 0; r < 8; ++r) { m[r] = -3.0e38f; l[r] = 0.0f; }

  const float sl = 0.125f * 1.44269504089f;  // SCALE * log2(e)
  _Float16* pP = ldsP + wave * 16 * KBLK;

  for (int j0 = 0; j0 < SEQ; j0 += KBLK) {
    __syncthreads();  // previous tile fully consumed before overwrite

    // --- stage K tile [64 keys][64 d]: contiguous 8KB, 2 x 16B per thread ---
    {
      const _Float16* src = kb + (size_t)j0 * DHEAD;
#pragma unroll
      for (int pass = 0; pass < 2; ++pass) {
        int off = (tid + pass * 256) * 8;   // halves
        copy16_g2l(ldsK + off, src + off);
      }
    }
    // --- stage V^T tile [64 d][64 keys]: 64 rows x 128B from stride-4KB rows -
    {
#pragma unroll
      for (int pass = 0; pass < 2; ++pass) {
        int q = tid + pass * 256;           // 0..511 16B-chunks
        int d = q >> 3;
        int c = (q & 7) * 8;                // halves within row
        copy16_g2l(ldsV + d * KBLK + c, vb + (size_t)d * SEQ + j0 + c);
      }
    }
    wait_async_copies();
    __syncthreads();

    // --- S = Q K^T  (16 x 64), K-dim 64 chained over two WMMAs -------------
    v8f s[4];
#pragma unroll
    for (int nt = 0; nt < 4; ++nt) {
#pragma unroll
      for (int r = 0; r < 8; ++r) s[nt][r] = 0.0f;
#pragma unroll
      for (int kt = 0; kt < 2; ++kt) {
        v16h kf = load_b_frag(ldsK + (nt * 16 + lidx) * DHEAD + kt * 32 +
                              half * 16);
        s[nt] = wmma16(qf[kt], kf, s[nt]);
      }
    }
#pragma unroll
    for (int nt = 0; nt < 4; ++nt)
#pragma unroll
      for (int r = 0; r < 8; ++r) s[nt][r] *= sl;  // log2 domain

    // --- online softmax: row max / sum over 64 keys (16-lane C groups) -----
    float rmax[8];
#pragma unroll
    for (int r = 0; r < 8; ++r) {
      float v = fmaxf(fmaxf(s[0][r], s[1][r]), fmaxf(s[2][r], s[3][r]));
      v = fmaxf(v, __shfl_xor(v, 1, 32));
      v = fmaxf(v, __shfl_xor(v, 2, 32));
      v = fmaxf(v, __shfl_xor(v, 4, 32));
      v = fmaxf(v, __shfl_xor(v, 8, 32));
      rmax[r] = v;
    }
    float alpha[8];
#pragma unroll
    for (int r = 0; r < 8; ++r) {
      float mn = fmaxf(m[r], rmax[r]);
      alpha[r] = exp2f(m[r] - mn);
      m[r] = mn;
    }
#pragma unroll
    for (int r = 0; r < 8; ++r) {
      float acc = 0.0f;
#pragma unroll
      for (int nt = 0; nt < 4; ++nt) {
        float p = exp2f(s[nt][r] - m[r]);
        s[nt][r] = p;
        acc += p;
      }
      acc += __shfl_xor(acc, 1, 32);
      acc += __shfl_xor(acc, 2, 32);
      acc += __shfl_xor(acc, 4, 32);
      acc += __shfl_xor(acc, 8, 32);
      l[r] = l[r] * alpha[r] + acc;
    }
#pragma unroll
    for (int t = 0; t < 4; ++t)
#pragma unroll
      for (int r = 0; r < 8; ++r) o[t][r] *= alpha[r];

    // --- C-layout -> A-layout redistribution of P via per-wave LDS tile ----
#pragma unroll
    for (int nt = 0; nt < 4; ++nt)
#pragma unroll
      for (int r = 0; r < 8; ++r)
        pP[(r + half * 8) * KBLK + nt * 16 + lidx] = (_Float16)s[nt][r];
    asm volatile("s_wait_dscnt 0" ::: "memory");
    v16h pa[2];
#pragma unroll
    for (int c = 0; c < 2; ++c)
      pa[c] = load_a_frag(pP + lidx * KBLK, c * 32 + half * 16,
                          c * 32 + half * 16 + 8);

    // --- O += P * V  (V^T tile in LDS, contiguous in key index) ------------
#pragma unroll
    for (int t = 0; t < 4; ++t) {
#pragma unroll
      for (int c = 0; c < 2; ++c) {
        v16h vf = load_b_frag(ldsV + (t * 16 + lidx) * KBLK + c * 32 +
                              half * 16);
        o[t] = wmma16(pa[c], vf, o[t]);
      }
    }
  }

  // normalize + store to token-major [b][n][h*64+d] f16
#pragma unroll
  for (int r = 0; r < 8; ++r) l[r] = 1.0f / l[r];
#pragma unroll
  for (int t = 0; t < 4; ++t) {
#pragma unroll
    for (int r = 0; r < 8; ++r) {
      int n = qBase + r + half * 8;
      int d = t * 16 + lidx;
      oh[((size_t)(bb * SEQ + n)) * DMODEL + hh * DHEAD + d] =
          (_Float16)(o[t][r] * l[r]);
    }
  }
}

// ---------------- output projection GEMM (+bias, f32 out) -------------------
__global__ void out_gemm(const _Float16* __restrict__ oh,
                         const _Float16* __restrict__ wT,
                         const float* __restrict__ bias,
                         float* __restrict__ out) {
  const int lane = threadIdx.x & 31;
  const int wave = threadIdx.x >> 5;
  const int half = lane >> 4;
  const int lidx = lane & 15;
  const int mBase = blockIdx.x * 128 + wave * 16;
  const int nBase = blockIdx.y * 64;

  v8f acc[4];
#pragma unroll
  for (int t = 0; t < 4; ++t)
#pragma unroll
    for (int r = 0; r < 8; ++r) acc[t][r] = 0.0f;

  const _Float16* arow = oh + (size_t)(mBase + lidx) * DMODEL;
  for (int k0 = 0; k0 < DMODEL; k0 += 32) {
    v16h a = load_a_frag(arow, k0 + half * 16, k0 + half * 16 + 8);
#pragma unroll
    for (int t = 0; t < 4; ++t) {
      v16h bf = load_b_frag(wT + (size_t)(nBase + t * 16 + lidx) * DMODEL +
                            k0 + half * 16);
      acc[t] = wmma16(a, bf, acc[t]);
    }
  }
#pragma unroll
  for (int t = 0; t < 4; ++t) {
    int col = nBase + t * 16 + lidx;
    float bv = bias[col];
#pragma unroll
    for (int r = 0; r < 8; ++r) {
      int row = mBase + r + half * 8;
      out[(size_t)row * DMODEL + col] = acc[t][r] + bv;
    }
  }
}

// ---------------- launcher ---------------------------------------------------
extern "C" void kernel_launch(void* const* d_in, const int* in_sizes, int n_in,
                              void* d_out, int out_size, void* d_ws, size_t ws_size,
                              hipStream_t stream) {
  (void)in_sizes; (void)n_in; (void)out_size; (void)ws_size;
  const float* x     = (const float*)d_in[0];  // [4,2048,512]
  const float* w_qkv = (const float*)d_in[1];  // [512,1536]
  const float* w_out = (const float*)d_in[2];  // [512,512]
  const float* b_out = (const float*)d_in[3];  // [512]
  float* out = (float*)d_out;                  // [4,2048,512]

  // workspace carve-up (all chunks >= 256B aligned)
  char* ws = (char*)d_ws;
  _Float16* xh    = (_Float16*)ws; ws += (size_t)ROWS * DMODEL * 2;       // 8 MB
  _Float16* wqkvT = (_Float16*)ws; ws += (size_t)NQKV * DMODEL * 2;       // 1.5 MB
  _Float16* woutT = (_Float16*)ws; ws += (size_t)DMODEL * DMODEL * 2;     // 0.5 MB
  _Float16* qws   = (_Float16*)ws; ws += (size_t)ROWS * DMODEL * 2;       // 8 MB
  _Float16* kws   = (_Float16*)ws; ws += (size_t)ROWS * DMODEL * 2;       // 8 MB
  _Float16* vws   = (_Float16*)ws; ws += (size_t)ROWS * DMODEL * 2;       // 8 MB
  _Float16* ohalf = (_Float16*)ws;                                        // 8 MB

  cvt_f32_to_f16<<<(ROWS * DMODEL + 255) / 256, 256, 0, stream>>>(
      x, xh, ROWS * DMODEL);
  transpose_cvt<<<(DMODEL * NQKV + 255) / 256, 256, 0, stream>>>(
      w_qkv, wqkvT, DMODEL, NQKV);
  transpose_cvt<<<(DMODEL * DMODEL + 255) / 256, 256, 0, stream>>>(
      w_out, woutT, DMODEL, DMODEL);

  qkv_gemm<<<dim3(ROWS / 128, NQKV / 64), 256, 0, stream>>>(
      xh, wqkvT, qws, kws, vws);

  attn_kernel<<<BATCH * HEADS * (SEQ / 128), 256, 0, stream>>>(
      qws, kws, vws, ohalf);

  out_gemm<<<dim3(ROWS / 128, DMODEL / 64), 256, 0, stream>>>(
      ohalf, woutT, b_out, out);
}
